// SparseConv2d_32684701122963
// MI455X (gfx1250) — matrix operand order repaired and checked by reading, not compiled, not run on hardware
//
#include <hip/hip_runtime.h>

typedef __attribute__((ext_vector_type(16))) __bf16 v16bf;
typedef __attribute__((ext_vector_type(8)))  float  v8f;
typedef __attribute__((ext_vector_type(4)))  int    v4i;

union FragBF { uint4 q[2]; v16bf v; };

__device__ __forceinline__ unsigned short f2b(float f) {
    unsigned u = __float_as_uint(f);
    unsigned r = u + 0x7fffu + ((u >> 16) & 1u);   // round-to-nearest-even
    return (unsigned short)(r >> 16);
}

// ---------------- constants ----------------
// GEMM: M = O = 512, N = B*H*W = 100352, K = 9*512 (tap-major: k = tap*512 + c)
// Activations staged bf16 channel-last: xb[b][s][c]  (K contiguous in memory)
#define CIN   512
#define OCH   512
#define HH    56
#define WW    56
#define SPAT  3136            // 56*56
#define NTOT  100352          // 32*3136
#define KTOT  4608            // 9*512
#define BM    64
#define BN    128
#define BK    64
#define NIT   (KTOT / BK)     // 72; 8 iterations per tap (512/64)
#define APITCH 144            // 64 bf16 = 128B data + 16B pad (b128-aligned frags)
#define BPITCH 144
#define ABYTES (BM * APITCH)          //  9216
#define BBYTES (BN * BPITCH)          // 18432
#define STAGE  (ABYTES + BBYTES)      // 27648

// -------- CDNA5 async global->LDS copy (guarded; fallback = reg pipeline) ----
#if defined(__gfx1250__) && __has_builtin(__builtin_amdgcn_global_load_async_to_lds_b128)
#define HAS_ASYNC 1
typedef __attribute__((address_space(1))) v4i* g_v4i_p;
typedef __attribute__((address_space(3))) v4i* l_v4i_p;
#define ASYNC16(g, l, OFF)                                                     \
    __builtin_amdgcn_global_load_async_to_lds_b128(                            \
        (g_v4i_p)(v4i*)(void*)(const_cast<unsigned short*>(g)),                \
        (l_v4i_p)(v4i*)(void*)(l), (OFF), 0)
#if __has_builtin(__builtin_amdgcn_s_wait_asynccnt)
#define WAIT_ASYNC() __builtin_amdgcn_s_wait_asynccnt(0)
#else
#define WAIT_ASYNC() asm volatile("s_wait_asynccnt 0x0" ::: "memory")
#endif
#else
#define HAS_ASYNC 0
#endif

// ---------------- prologue kernels ----------------
__global__ void zero4(float4* __restrict__ p) {
    p[blockIdx.x * 256 + threadIdx.x] = make_float4(0.f, 0.f, 0.f, 0.f);
}

__global__ void scatter_w(const long long* __restrict__ idx,
                          const float* __restrict__ val,
                          float* __restrict__ dwf) {
    int i = blockIdx.x * 256 + threadIdx.x;        // 512*64*9 threads exactly
    int o   = i / 576;
    int rem = i - o * 576;
    int j   = rem / 9;
    int tap = rem - j * 9;
    int c = (int)idx[o * 64 + j];
    // tap-major dense layout: k = tap*512 + c ; duplicates sum like reference .add
    atomicAdd(&dwf[o * KTOT + tap * CIN + c], val[i]);
}

__global__ void cvt_w(const float4* __restrict__ src, ushort4* __restrict__ dst) {
    int i = blockIdx.x * 256 + threadIdx.x;
    float4 v = src[i];
    ushort4 r; r.x = f2b(v.x); r.y = f2b(v.y); r.z = f2b(v.z); r.w = f2b(v.w);
    dst[i] = r;
}

// NCHW f32 -> NHWC bf16 tiled transpose: xb[b][s][c] = bf16(x[b][c][s])
__global__ __launch_bounds__(256)
void transpose_x(const float* __restrict__ x, unsigned short* __restrict__ xb) {
    __shared__ float tile[32][33];
    const int s0 = blockIdx.x * 32;        // 98 tiles
    const int c0 = blockIdx.y * 32;        // 16 tiles
    const int b  = blockIdx.z;             // 32
    const int tx = threadIdx.x & 31;
    const int ty = threadIdx.x >> 5;       // 0..7
#pragma unroll
    for (int i = 0; i < 4; ++i)            // coalesced read along s
        tile[ty + 8 * i][tx] = x[(b * CIN + c0 + ty + 8 * i) * SPAT + s0 + tx];
    __syncthreads();
#pragma unroll
    for (int i = 0; i < 4; ++i)            // coalesced write along c
        xb[(b * SPAT + s0 + ty + 8 * i) * CIN + c0 + tx] = f2b(tile[tx][ty + 8 * i]);
}

// ---------------- main implicit-GEMM conv (bf16 WMMA) ----------------
__global__ __launch_bounds__(256, 2)
void conv_wmma(const unsigned short* __restrict__ xb,   // bf16 x  [32][3136][512] (NHWC)
               const unsigned short* __restrict__ wb,   // bf16 W  [512][4608] tap-major
               const unsigned short* __restrict__ zpad, // 64KB of zeros (halo source)
               float* __restrict__ out) {
    __shared__ __align__(16) unsigned char smem[2 * STAGE];

    const int t     = threadIdx.x;
    const int mbase = blockIdx.x * BM;     // 8 tiles
    const int nbase = blockIdx.y * BN;     // 784 tiles

    // ---- B-tile gather mapping: thread stages 2x16 channels of one n row ----
    const int nloc  = t & 127;
    const int kq    = t >> 7;              // 0/1: channel sub-chunks (kq*16, 32+kq*16)
    const int n     = nbase + nloc;
    const int bimg  = n / SPAT;
    const int s     = n - bimg * SPAT;
    const int h     = s / WW;
    const int w     = s - h * WW;
    const unsigned short* xrow = xb + (long)n * CIN;   // row (b, h, w), channel-contiguous

    // per-tap halo validity, computed once (9 bits)
    unsigned vmask = 0;
#pragma unroll
    for (int tap = 0; tap < 9; ++tap) {
        int dh = tap / 3 - 1, dw = tap % 3 - 1;
        if ((unsigned)(h + dh) < (unsigned)HH && (unsigned)(w + dw) < (unsigned)WW)
            vmask |= 1u << tap;
    }

    // ---- A-tile mapping: thread copies 16 consecutive k of one m row ----
    const int am = t >> 2;                 // 0..63
    const int ac = t & 3;                  // chunk of 16 bf16
    const unsigned short* aptr = wb + (unsigned)(mbase + am) * KTOT + ac * 16;

    // ---- compute mapping ----
    const int lane  = t & 31;
    const int wv    = t >> 5;              // 8 waves
    const int wm    = (wv & 1) * 32;       // 2 x 4 wave grid over 64 x 128
    const int wn    = (wv >> 1) * 32;
    const int lhalf = lane >> 4;
    const int l15   = lane & 15;

    v8f acc[2][2] = {};

    auto compute = [&](int buf) {
        const unsigned char* base = smem + buf * STAGE;
#pragma unroll
        for (int kh = 0; kh < 2; ++kh) {   // two 32-wide k-halves per stage
            const unsigned char* ab = base + (wm + l15) * APITCH + kh * 64 + lhalf * 32;
            const unsigned char* bb = base + ABYTES + (wn + l15) * BPITCH + kh * 64 + lhalf * 32;
            FragBF a0, a1, b0, b1;
            a0.q[0] = *(const uint4*)(ab);
            a0.q[1] = *(const uint4*)(ab + 16);
            a1.q[0] = *(const uint4*)(ab + 16 * APITCH);
            a1.q[1] = *(const uint4*)(ab + 16 * APITCH + 16);
            b0.q[0] = *(const uint4*)(bb);
            b0.q[1] = *(const uint4*)(bb + 16);
            b1.q[0] = *(const uint4*)(bb + 16 * BPITCH);
            b1.q[1] = *(const uint4*)(bb + 16 * BPITCH + 16);
            acc[0][0] = __builtin_amdgcn_wmma_f32_16x16x32_bf16(false, a0.v, false, b0.v,
                                                                (short)0, acc[0][0], false, false);
            acc[0][1] = __builtin_amdgcn_wmma_f32_16x16x32_bf16(false, a0.v, false, b1.v,
                                                                (short)0, acc[0][1], false, false);
            acc[1][0] = __builtin_amdgcn_wmma_f32_16x16x32_bf16(false, a1.v, false, b0.v,
                                                                (short)0, acc[1][0], false, false);
            acc[1][1] = __builtin_amdgcn_wmma_f32_16x16x32_bf16(false, a1.v, false, b1.v,
                                                                (short)0, acc[1][1], false, false);
        }
    };

#if HAS_ASYNC
    // -------- async global->LDS pipeline (ASYNCcnt), no staging registers ----
    auto issue_async = [&](int it, int buf) {
        unsigned char* base = smem + buf * STAGE;
        const int kk = it * BK;
        // A tile: 32B contiguous per thread
        const unsigned short* asrc = aptr + kk;
        unsigned char*        adst = base + am * APITCH + ac * 32;
        ASYNC16(asrc, adst, 0);
        ASYNC16(asrc, adst, 16);
        // B tile: 2 x 32B contiguous chunks per thread (tap-uniform shift)
        const int tap  = it >> 3;
        const int dh   = tap / 3 - 1;
        const int dw   = tap - (tap / 3) * 3 - 1;
        const int soff = dh * WW + dw;
        const int c0   = (it & 7) * BK + kq * 16;
        const bool ok  = (vmask >> tap) & 1u;
        const unsigned short* bsrc = ok ? (xrow + soff * CIN + c0) : zpad;
        unsigned char*        bdst = base + ABYTES + nloc * BPITCH + kq * 32;
        ASYNC16(bsrc, bdst, 0);    // channels c0..c0+15   -> k bytes kq*32..
        ASYNC16(bsrc, bdst, 16);
        ASYNC16(bsrc, bdst, 64);   // channels c0+32..+47  -> k bytes 64+kq*32..
        ASYNC16(bsrc, bdst, 80);
    };

    issue_async(0, 0);
    WAIT_ASYNC();
    __syncthreads();
#pragma unroll 2
    for (int it = 0; it < NIT; ++it) {
        const int buf = it & 1;
        if (it + 1 < NIT) issue_async(it + 1, buf ^ 1);  // DMA overlaps WMMA
        compute(buf);
        WAIT_ASYNC();
        __syncthreads();
    }
#else
    // -------- fallback: global -> regs -> LDS software pipeline ----
    uint4 areg0, areg1, breg0, breg1, breg2, breg3;
    auto load_regs = [&](int it) {
        const int kk = it * BK;
        areg0 = *(const uint4*)(aptr + kk);
        areg1 = *(const uint4*)(aptr + kk + 8);
        const int tap  = it >> 3;
        const int dh   = tap / 3 - 1;
        const int dw   = tap - (tap / 3) * 3 - 1;
        const int soff = dh * WW + dw;
        const int c0   = (it & 7) * BK + kq * 16;
        const bool ok  = (vmask >> tap) & 1u;
        const uint4* q  = (const uint4*)(xrow + soff * CIN + c0);
        const uint4* q2 = (const uint4*)(xrow + soff * CIN + c0 + 32);
        uint4 b0 = {0,0,0,0}, b1 = {0,0,0,0}, b2 = {0,0,0,0}, b3 = {0,0,0,0};
        if (ok) { b0 = q[0]; b1 = q[1]; b2 = q2[0]; b3 = q2[1]; }
        breg0 = b0; breg1 = b1; breg2 = b2; breg3 = b3;
    };
    auto store_lds = [&](int buf) {
        unsigned char* base = smem + buf * STAGE;
        unsigned char* aa = base + am * APITCH + ac * 32;
        *(uint4*)(aa)      = areg0;
        *(uint4*)(aa + 16) = areg1;
        unsigned char* bb = base + ABYTES + nloc * BPITCH + kq * 32;
        *(uint4*)(bb)      = breg0;
        *(uint4*)(bb + 16) = breg1;
        *(uint4*)(bb + 64) = breg2;
        *(uint4*)(bb + 80) = breg3;
    };
    load_regs(0);
    store_lds(0);
    __syncthreads();
#pragma unroll 2
    for (int it = 0; it < NIT; ++it) {
        const int buf = it & 1;
        if (it + 1 < NIT) load_regs(it + 1);
        compute(buf);
        if (it + 1 < NIT) store_lds(buf ^ 1);
        __syncthreads();
    }
#endif

    // ---- epilogue: D layout (VGPR r -> M = r + 8*half, lane%16 -> N col) ----
#pragma unroll
    for (int fm = 0; fm < 2; ++fm) {
#pragma unroll
        for (int fn = 0; fn < 2; ++fn) {
            int ncol = nbase + wn + fn * 16 + l15;
            int b2 = ncol / SPAT;
            int s2 = ncol - b2 * SPAT;
#pragma unroll
            for (int r = 0; r < 8; ++r) {
                int o = mbase + wm + fm * 16 + r + 8 * lhalf;
                __builtin_nontemporal_store(acc[fm][fn][r],
                    out + (unsigned)(b2 * OCH + o) * SPAT + s2);
            }
        }
    }
}

// ---------------- host launch ----------------
extern "C" void kernel_launch(void* const* d_in, const int* in_sizes, int n_in,
                              void* d_out, int out_size, void* d_ws, size_t ws_size,
                              hipStream_t stream) {
    const float*     x    = (const float*)d_in[0];
    const long long* widx = (const long long*)d_in[1];   // int64 per reference
    const float*     wval = (const float*)d_in[2];
    float* out = (float*)d_out;

    char* ws = (char*)d_ws;
    float*          dwf  = (float*)ws;                         //  9,437,184 B dense f32 W
    unsigned short* dwb  = (unsigned short*)(ws + 9437184);    //  4,718,592 B bf16 W
    unsigned short* zpad = (unsigned short*)(ws + 14155776);   //     65,536 B zeros
    unsigned short* xb   = (unsigned short*)(ws + 14221312);   // 102,760,448 B NHWC bf16

    // 512*4608 floats = 589824 float4 ; zero pad = 4096 float4
    zero4    <<<2304, 256, 0, stream>>>((float4*)dwf);
    zero4    <<<16,   256, 0, stream>>>((float4*)zpad);
    // 512*64*9 = 294912 scatter-adds
    scatter_w<<<1152, 256, 0, stream>>>(widx, wval, dwf);
    cvt_w    <<<2304, 256, 0, stream>>>((const float4*)dwf, (ushort4*)dwb);
    // NCHW f32 -> NHWC bf16
    dim3 tgrid(SPAT / 32, CIN / 32, 32);  // 98 x 16 x 32
    transpose_x<<<tgrid, 256, 0, stream>>>(x, xb);

    dim3 grid(OCH / BM, NTOT / BN);       // 8 x 784
    conv_wmma<<<grid, 256, 0, stream>>>(xb, dwb, zpad, out);
}
